// CustomTransformerClassifier_51049981281149
// MI455X (gfx1250) — matrix operand order repaired and checked by reading, not compiled, not run on hardware
//
#include <hip/hip_runtime.h>

// ---------------------------------------------------------------------------
// Types for CDNA5 WMMA (gfx1250, wave32)
// ---------------------------------------------------------------------------
typedef __bf16 bf16x8  __attribute__((ext_vector_type(8)));
typedef __bf16 bf16x16 __attribute__((ext_vector_type(16)));
typedef float  f32x8   __attribute__((ext_vector_type(8)));
typedef int    v4i     __attribute__((ext_vector_type(4)));

static constexpr int Dm   = 768;
static constexpr int Hh   = 12;
static constexpr int DK   = 64;
static constexpr int DFF  = 3072;
static constexpr int Ll   = 12;
static constexpr int NC   = 100;
static constexpr int Ss   = 512;
static constexpr int Bb   = 16;
static constexpr int Mrow = Bb * Ss;        // 8192
static constexpr int RELN = 2 * Ss - 1;     // 1023

__device__ __forceinline__ f32x8 wmma_bf16(bf16x16 a, bf16x16 b, f32x8 c) {
  // D = A(16x32 bf16) * B(32x16 bf16) + C(16x16 f32)
  return __builtin_amdgcn_wmma_f32_16x16x32_bf16(
      /*neg_a=*/false, a, /*neg_b=*/false, b,
      /*c_mod=*/(short)0, c, /*reuse_a=*/false, /*reuse_b=*/false);
}

__device__ __forceinline__ bf16x16 cat8(bf16x8 lo, bf16x8 hi) {
  bf16x16 r;
#pragma unroll
  for (int i = 0; i < 8; ++i) { r[i] = lo[i]; r[i + 8] = hi[i]; }
  return r;
}

__device__ __forceinline__ f32x8 zero8() {
  f32x8 z = {0.f, 0.f, 0.f, 0.f, 0.f, 0.f, 0.f, 0.f};
  return z;
}

// ---------------------------------------------------------------------------
// Async global -> LDS copy path (CDNA5 GLOBAL_LOAD_ASYNC_TO_LDS_B128,
// tracked by ASYNCcnt). Builtin signature (from hipcc diagnostic): generic
// v4i* pointers (global src, LDS dst) + two int immediates (offset, cpol).
// Guarded so the file still compiles on toolchains without the builtin.
// ---------------------------------------------------------------------------
#if __has_builtin(__builtin_amdgcn_global_load_async_to_lds_b128)
#define HAVE_ASYNC_LDS 1
#else
#define HAVE_ASYNC_LDS 0
#endif

__device__ __forceinline__ void cp_glb_to_lds_b128(__bf16* lds, const __bf16* g) {
#if HAVE_ASYNC_LDS
  __builtin_amdgcn_global_load_async_to_lds_b128(
      (v4i*)(void*)g, (v4i*)(void*)lds, /*offset=*/0, /*cpol=*/0);
#else
  *(bf16x8*)lds = *(const bf16x8*)g;
#endif
}

__device__ __forceinline__ void cp_async_wait() {
#if HAVE_ASYNC_LDS
#if __has_builtin(__builtin_amdgcn_s_wait_asynccnt)
  __builtin_amdgcn_s_wait_asynccnt(0);
#else
  asm volatile("s_wait_asynccnt 0x0" ::: "memory");
#endif
#endif
}

// ---------------------------------------------------------------------------
// Lane-xor reductions via ds_swizzle (group-of-32 mode: and=0x1F, xor=MASK).
// ---------------------------------------------------------------------------
template <int MASK>
__device__ __forceinline__ float lane_xor(float v) {
  return __int_as_float(
      __builtin_amdgcn_ds_swizzle(__float_as_int(v), (MASK << 10) | 0x1f));
}
__device__ __forceinline__ float redmax16(float v) {
  v = fmaxf(v, lane_xor<1>(v));
  v = fmaxf(v, lane_xor<2>(v));
  v = fmaxf(v, lane_xor<4>(v));
  v = fmaxf(v, lane_xor<8>(v));
  return v;
}
__device__ __forceinline__ float redsum16(float v) {
  v += lane_xor<1>(v); v += lane_xor<2>(v);
  v += lane_xor<4>(v); v += lane_xor<8>(v);
  return v;
}
__device__ __forceinline__ float redsum32(float v) {
  v += lane_xor<1>(v); v += lane_xor<2>(v); v += lane_xor<4>(v);
  v += lane_xor<8>(v); v += lane_xor<16>(v);
  return v;
}

// ---------------------------------------------------------------------------
// Generic bf16 WMMA GEMM:  C[M,N] = A[M,K] * Bt[N,K]^T (+bias) (+relu)
// A row-major bf16, Bt row-major bf16 (B pre-transposed). Block tile 128x128,
// 8 waves (4Mx2N), wave tile 32x64. Double-buffered LDS with async staging.
// ---------------------------------------------------------------------------
template <bool RELU>
__global__ __launch_bounds__(256) void gemm_bf16_wmma(
    const __bf16* __restrict__ A, const __bf16* __restrict__ Bt,
    const float* __restrict__ bias, float* __restrict__ Cf,
    __bf16* __restrict__ Cb, int M, int N, int K) {
  __shared__ __align__(16) __bf16 As[2][128 * 32];
  __shared__ __align__(16) __bf16 Bs[2][128 * 32];

  const int ntiles = (N + 127) >> 7;
  const int m0 = (blockIdx.x / ntiles) * 128;
  const int n0 = (blockIdx.x % ntiles) * 128;
  const int tid  = threadIdx.x;
  const int wave = tid >> 5, lane = tid & 31;
  const int wm = wave & 3, wn = wave >> 2;   // 4 x 2 wave grid
  const int lh = lane >> 4, ll = lane & 15;  // half / lane-in-half
  const int rr0 = tid >> 2;                  // staging row 0..63
  const int cc  = (tid & 3) << 3;            // staging col 0..24 step 8

  auto stage = [&](int buf, int k0) {
#pragma unroll
    for (int p = 0; p < 2; ++p) {
      const int r = p * 64 + rr0;
      int ga = m0 + r; if (ga >= M) ga = M - 1;
      int gb = n0 + r; if (gb >= N) gb = N - 1;
      cp_glb_to_lds_b128(&As[buf][r * 32 + cc], A  + (size_t)ga * K + k0 + cc);
      cp_glb_to_lds_b128(&Bs[buf][r * 32 + cc], Bt + (size_t)gb * K + k0 + cc);
    }
  };

  f32x8 acc[2][4];
#pragma unroll
  for (int i = 0; i < 2; ++i)
#pragma unroll
    for (int j = 0; j < 4; ++j) acc[i][j] = zero8();

  stage(0, 0);
  int buf = 0;
  for (int k0 = 0; k0 < K; k0 += 32) {
    cp_async_wait();
    __syncthreads();
    if (k0 + 32 < K) stage(buf ^ 1, k0 + 32);  // overlap next slab with WMMA

    // A fragments: 16x32 bf16, lane%16 = row; lanes>=16 hold odd K-octets.
    bf16x16 afr[2];
#pragma unroll
    for (int i = 0; i < 2; ++i) {
      const __bf16* p = &As[buf][(wm * 32 + i * 16 + ll) * 32 + lh * 8];
      afr[i] = cat8(*(const bf16x8*)p, *(const bf16x8*)(p + 16));
    }
    // B fragments: 32x16 bf16, lane%16 = column (row of Bt); lanes>=16 hold odd K-octets.
    bf16x16 bfr[4];
#pragma unroll
    for (int j = 0; j < 4; ++j) {
      const __bf16* p = &Bs[buf][(wn * 64 + j * 16 + ll) * 32 + lh * 8];
      bfr[j] = cat8(*(const bf16x8*)p, *(const bf16x8*)(p + 16));
    }
#pragma unroll
    for (int i = 0; i < 2; ++i)
#pragma unroll
      for (int j = 0; j < 4; ++j) acc[i][j] = wmma_bf16(afr[i], bfr[j], acc[i][j]);
    buf ^= 1;
  }

  // Epilogue. C layout: lane%16 = N, row = 8*(lane/16) + vgpr.
#pragma unroll
  for (int i = 0; i < 2; ++i) {
#pragma unroll
    for (int j = 0; j < 4; ++j) {
      const int col = n0 + wn * 64 + j * 16 + ll;
      const int rb  = m0 + wm * 32 + i * 16 + lh * 8;
      const float bv = (bias != nullptr && col < N) ? bias[col] : 0.f;
#pragma unroll
      for (int r = 0; r < 8; ++r) {
        const int gr = rb + r;
        if (gr < M && col < N) {
          float v = acc[i][j][r] + bv;
          if (RELU) v = fmaxf(v, 0.f);
          if (Cf) Cf[(size_t)gr * N + col] = v;
          if (Cb) Cb[(size_t)gr * N + col] = (__bf16)v;
        }
      }
    }
  }
}

// ---------------------------------------------------------------------------
// Fused multi-head attention with relative position bias (flash-style).
// Grid: B * H * (S/128); block = 256 threads = 8 waves; wave owns 16 q-rows.
// Q/K/V/Ctx are bf16 [B, S, H, DK] (== [M, 768]).
// ---------------------------------------------------------------------------
__global__ __launch_bounds__(256) void attn_wmma(
    const __bf16* __restrict__ Q, const __bf16* __restrict__ Km,
    const __bf16* __restrict__ Vm, const float* __restrict__ relb,  // [1023,H]
    __bf16* __restrict__ Ctx) {
  __shared__ __align__(16) float  bias_tab[RELN];
  __shared__ __align__(16) __bf16 Ks[32 * 64];    // K tile  [kv][d]
  __shared__ __align__(16) __bf16 Vt[64 * 32];    // V tile  [d][kv] (transposed)
  __shared__ __align__(16) __bf16 Ps[8][16 * 32]; // per-wave P tile [q][kv]

  const int qb = blockIdx.x & 3;
  const int h  = (blockIdx.x >> 2) % Hh;
  const int b  = blockIdx.x / (4 * Hh);
  const int tid  = threadIdx.x;
  const int wave = tid >> 5, lane = tid & 31;
  const int lh = lane >> 4, ll = lane & 15;
  const int q0 = qb * 128 + wave * 16;

  for (int i = tid; i < RELN; i += 256) bias_tab[i] = relb[(size_t)i * Hh + h];

  // Q fragments (A layout): 16 rows x 64 deep = two 16x32 frags.
  bf16x16 aq[2];
  {
    const __bf16* qr = Q + ((size_t)(b * Ss + q0 + ll)) * Dm + h * DK;
#pragma unroll
    for (int f = 0; f < 2; ++f)
      aq[f] = cat8(*(const bf16x8*)(qr + f * 32 + lh * 8),
                   *(const bf16x8*)(qr + f * 32 + 16 + lh * 8));
  }

  f32x8 cacc[4];
#pragma unroll
  for (int t = 0; t < 4; ++t) cacc[t] = zero8();
  float m_run[8], l_run[8];
#pragma unroll
  for (int r = 0; r < 8; ++r) { m_run[r] = -1e30f; l_run[r] = 0.f; }

  const int kvr = tid >> 3;        // 0..31
  const int ck  = (tid & 7) << 3;  // 0..56 step 8

  for (int it = 0; it < Ss / 32; ++it) {
    const int kv0 = it * 32;
    __syncthreads();
    {  // stage: K row-major (async global->LDS), V transposed (sync).
      const size_t src = ((size_t)(b * Ss + kv0 + kvr)) * Dm + h * DK + ck;
      cp_glb_to_lds_b128(Ks + kvr * 64 + ck, Km + src);
      const bf16x8 vv = *(const bf16x8*)(Vm + src);
#pragma unroll
      for (int e = 0; e < 8; ++e) Vt[(ck + e) * 32 + kvr] = vv[e];
    }
    cp_async_wait();
    __syncthreads();

    // Scores: two 16x16 tiles (kv columns kv0..kv0+31).
    float pv[2][8];
#pragma unroll
    for (int jn = 0; jn < 2; ++jn) {
      f32x8 s = zero8();
#pragma unroll
      for (int f = 0; f < 2; ++f) {
        const __bf16* kr = Ks + (jn * 16 + ll) * 64 + f * 32 + lh * 8;
        const bf16x16 bk = cat8(*(const bf16x8*)kr, *(const bf16x8*)(kr + 16));
        s = wmma_bf16(aq[f], bk, s);
      }
#pragma unroll
      for (int r = 0; r < 8; ++r) {
        const int qi  = q0 + lh * 8 + r;
        const int kj  = kv0 + jn * 16 + ll;
        int idx = kj - qi + (Ss - 1);
        idx = idx < 0 ? 0 : (idx > RELN - 1 ? RELN - 1 : idx);
        pv[jn][r] = s[r] * 0.125f + bias_tab[idx];
      }
    }

    // Online softmax (row reductions across the 16 lanes sharing a row set).
#pragma unroll
    for (int r = 0; r < 8; ++r) {
      const float mx   = redmax16(fmaxf(pv[0][r], pv[1][r]));
      const float mnew = fmaxf(m_run[r], mx);
      const float scl  = __expf(m_run[r] - mnew);
      m_run[r] = mnew;
      l_run[r] *= scl;
#pragma unroll
      for (int t = 0; t < 4; ++t) cacc[t][r] *= scl;
      const float p0 = __expf(pv[0][r] - mnew);
      const float p1 = __expf(pv[1][r] - mnew);
      pv[0][r] = p0; pv[1][r] = p1;
      l_run[r] += redsum16(p0 + p1);
    }

    // C-layout -> A-layout relayout of P through per-wave LDS slab.
    __bf16* pw = &Ps[wave][0];
#pragma unroll
    for (int jn = 0; jn < 2; ++jn)
#pragma unroll
      for (int r = 0; r < 8; ++r)
        pw[(lh * 8 + r) * 32 + jn * 16 + ll] = (__bf16)pv[jn][r];

    // ctx += P(16x32) x V(32x64)
    {
      const __bf16* pr = pw + ll * 32 + lh * 8;
      const bf16x16 ap = cat8(*(const bf16x8*)pr, *(const bf16x8*)(pr + 16));
#pragma unroll
      for (int nt = 0; nt < 4; ++nt) {
        const __bf16* vr = Vt + (nt * 16 + ll) * 32 + lh * 8;
        const bf16x16 bv = cat8(*(const bf16x8*)vr, *(const bf16x8*)(vr + 16));
        cacc[nt] = wmma_bf16(ap, bv, cacc[nt]);
      }
    }
  }

  // Normalize and store ctx (bf16, [B,S,H,DK]).
#pragma unroll
  for (int r = 0; r < 8; ++r) {
    const float inv = l_run[r] > 0.f ? 1.f / l_run[r] : 0.f;
    const int qi = q0 + lh * 8 + r;
    __bf16* crow = Ctx + ((size_t)(b * Ss + qi)) * Dm + h * DK;
#pragma unroll
    for (int nt = 0; nt < 4; ++nt) crow[nt * 16 + ll] = (__bf16)(cacc[nt][r] * inv);
  }
}

// ---------------------------------------------------------------------------
// Residual add + LayerNorm: Y = LN(X + R) * g + b ; writes f32 + bf16.
// One block (256 thr) per row of 768.
// ---------------------------------------------------------------------------
__device__ __forceinline__ float block_sum256(float v, float* red) {
  v = redsum32(v);
  const int wave = threadIdx.x >> 5, lane = threadIdx.x & 31;
  __syncthreads();
  if (lane == 0) red[wave] = v;
  __syncthreads();
  float t = red[0];
#pragma unroll
  for (int w = 1; w < 8; ++w) t += red[w];
  return t;
}

__global__ __launch_bounds__(256) void add_ln_kernel(
    const float* __restrict__ X, const float* __restrict__ R,
    const float* __restrict__ g, const float* __restrict__ bb,
    float* __restrict__ Yf, __bf16* __restrict__ Yb) {
  __shared__ float red[8];
  const int row = blockIdx.x, tid = threadIdx.x;
  const float* x = X + (size_t)row * Dm;
  const float* rr = R + (size_t)row * Dm;
  float v[3]; float s = 0.f;
#pragma unroll
  for (int i = 0; i < 3; ++i) { const int c = tid + i * 256; v[i] = x[c] + rr[c]; s += v[i]; }
  const float mean = block_sum256(s, red) * (1.f / Dm);
  float ss = 0.f;
#pragma unroll
  for (int i = 0; i < 3; ++i) { const float d = v[i] - mean; ss += d * d; }
  const float inv = rsqrtf(block_sum256(ss, red) * (1.f / Dm) + 1e-5f);
#pragma unroll
  for (int i = 0; i < 3; ++i) {
    const int c = tid + i * 256;
    const float o = (v[i] - mean) * inv * g[c] + bb[c];
    Yf[(size_t)row * Dm + c] = o;
    Yb[(size_t)row * Dm + c] = (__bf16)o;
  }
}

// ---------------------------------------------------------------------------
// Embedding + sinusoidal positional encoding.
// ---------------------------------------------------------------------------
__global__ __launch_bounds__(256) void embed_kernel(
    const int* __restrict__ ids, const float* __restrict__ emb,
    float* __restrict__ Xf, __bf16* __restrict__ Xb) {
  const int idx = blockIdx.x * 256 + threadIdx.x;
  if (idx >= Mrow * Dm) return;
  const int m = idx / Dm, d = idx - m * Dm;
  const int s = m % Ss;
  const int tok = ids[m];
  const float div = __expf((float)(d & ~1) * (-9.210340371976184f / (float)Dm));
  const float ang = (float)s * div;
  const float pe = (d & 1) ? __cosf(ang) : __sinf(ang);
  const float v = emb[(size_t)tok * Dm + d] + pe;
  Xf[idx] = v;
  Xb[idx] = (__bf16)v;
}

// Mean pool over sequence.
__global__ __launch_bounds__(256) void pool_kernel(
    const float* __restrict__ Xf, float* __restrict__ Pf, __bf16* __restrict__ Pb) {
  const int idx = blockIdx.x * 256 + threadIdx.x;
  if (idx >= Bb * Dm) return;
  const int b = idx / Dm, d = idx - b * Dm;
  float s = 0.f;
  for (int t = 0; t < Ss; ++t) s += Xf[((size_t)(b * Ss + t)) * Dm + d];
  s *= (1.f / (float)Ss);
  Pf[idx] = s;
  Pb[idx] = (__bf16)s;
}

// Convert fp32 W[K,N] -> bf16 transposed Wt[N,K].
__global__ __launch_bounds__(256) void wconv_kernel(
    const float* __restrict__ W, __bf16* __restrict__ Wt, int K, int N) {
  const int idx = blockIdx.x * 256 + threadIdx.x;
  if (idx >= K * N) return;
  const int n = idx / K, k = idx - n * K;
  Wt[(size_t)n * K + k] = (__bf16)W[(size_t)k * N + n];
}

// ---------------------------------------------------------------------------
// Host orchestration
// ---------------------------------------------------------------------------
extern "C" void kernel_launch(void* const* d_in, const int* in_sizes, int n_in,
                              void* d_out, int out_size, void* d_ws, size_t ws_size,
                              hipStream_t stream) {
  const int*   ids  = (const int*)d_in[0];
  const float* emb  = (const float*)d_in[1];
  const float* wq   = (const float*)d_in[2];  const float* bq = (const float*)d_in[3];
  const float* wk   = (const float*)d_in[4];  const float* bk = (const float*)d_in[5];
  const float* wv   = (const float*)d_in[6];  const float* bv = (const float*)d_in[7];
  const float* wo   = (const float*)d_in[8];  const float* bo = (const float*)d_in[9];
  const float* relb = (const float*)d_in[10];
  const float* w1   = (const float*)d_in[11]; const float* b1 = (const float*)d_in[12];
  const float* w2   = (const float*)d_in[13]; const float* b2 = (const float*)d_in[14];
  const float* ln1g = (const float*)d_in[15]; const float* ln1b = (const float*)d_in[16];
  const float* ln2g = (const float*)d_in[17]; const float* ln2b = (const float*)d_in[18];
  const float* cw1  = (const float*)d_in[19]; const float* cb1 = (const float*)d_in[20];
  const float* cw2  = (const float*)d_in[21]; const float* cb2 = (const float*)d_in[22];

  char* cur = (char*)d_ws;
  auto alloc = [&](size_t bytes) -> void* {
    void* r = (void*)cur;
    cur += (bytes + 255) & ~(size_t)255;
    return r;
  };

  __bf16* Wqt = (__bf16*)alloc((size_t)Dm * Dm * 2);
  __bf16* Wkt = (__bf16*)alloc((size_t)Dm * Dm * 2);
  __bf16* Wvt = (__bf16*)alloc((size_t)Dm * Dm * 2);
  __bf16* Wot = (__bf16*)alloc((size_t)Dm * Dm * 2);
  __bf16* W1t = (__bf16*)alloc((size_t)Dm * DFF * 2);
  __bf16* W2t = (__bf16*)alloc((size_t)DFF * Dm * 2);
  __bf16* C1t = (__bf16*)alloc((size_t)(Dm / 2) * Dm * 2);
  __bf16* C2t = (__bf16*)alloc((size_t)NC * (Dm / 2) * 2);

  float*  xf  = (float*)alloc((size_t)Mrow * Dm * 4);
  __bf16* xb  = (__bf16*)alloc((size_t)Mrow * Dm * 2);
  __bf16* qB  = (__bf16*)alloc((size_t)Mrow * Dm * 2);
  __bf16* kB  = (__bf16*)alloc((size_t)Mrow * Dm * 2);
  __bf16* vB  = (__bf16*)alloc((size_t)Mrow * Dm * 2);
  __bf16* ctxB= (__bf16*)alloc((size_t)Mrow * Dm * 2);
  float*  aoF = (float*)alloc((size_t)Mrow * Dm * 4);
  float*  yF  = (float*)alloc((size_t)Mrow * Dm * 4);
  __bf16* yB  = (__bf16*)alloc((size_t)Mrow * Dm * 2);
  __bf16* hB  = (__bf16*)alloc((size_t)Mrow * DFF * 2);
  float*  ffF = (float*)alloc((size_t)Mrow * Dm * 4);
  float*  plF = (float*)alloc((size_t)Bb * Dm * 4);
  __bf16* plB = (__bf16*)alloc((size_t)Bb * Dm * 2);
  __bf16* h1B = (__bf16*)alloc((size_t)Bb * (Dm / 2) * 2);

  auto gemm_grid = [](int M, int N) { return ((M + 127) / 128) * ((N + 127) / 128); };

  // Embedding + positional encoding
  embed_kernel<<<(Mrow * Dm + 255) / 256, 256, 0, stream>>>(ids, emb, xf, xb);

  for (int l = 0; l < Ll; ++l) {
    const size_t oDD = (size_t)l * Dm * Dm;
    wconv_kernel<<<(Dm * Dm + 255) / 256, 256, 0, stream>>>(wq + oDD, Wqt, Dm, Dm);
    wconv_kernel<<<(Dm * Dm + 255) / 256, 256, 0, stream>>>(wk + oDD, Wkt, Dm, Dm);
    wconv_kernel<<<(Dm * Dm + 255) / 256, 256, 0, stream>>>(wv + oDD, Wvt, Dm, Dm);
    wconv_kernel<<<(Dm * Dm + 255) / 256, 256, 0, stream>>>(wo + oDD, Wot, Dm, Dm);
    wconv_kernel<<<(Dm * DFF + 255) / 256, 256, 0, stream>>>(w1 + (size_t)l * Dm * DFF, W1t, Dm, DFF);
    wconv_kernel<<<(DFF * Dm + 255) / 256, 256, 0, stream>>>(w2 + (size_t)l * DFF * Dm, W2t, DFF, Dm);

    // Q, K, V projections (bf16 out)
    gemm_bf16_wmma<false><<<gemm_grid(Mrow, Dm), 256, 0, stream>>>(
        xb, Wqt, bq + (size_t)l * Dm, nullptr, qB, Mrow, Dm, Dm);
    gemm_bf16_wmma<false><<<gemm_grid(Mrow, Dm), 256, 0, stream>>>(
        xb, Wkt, bk + (size_t)l * Dm, nullptr, kB, Mrow, Dm, Dm);
    gemm_bf16_wmma<false><<<gemm_grid(Mrow, Dm), 256, 0, stream>>>(
        xb, Wvt, bv + (size_t)l * Dm, nullptr, vB, Mrow, Dm, Dm);

    // Fused attention with relative position bias
    attn_wmma<<<Bb * Hh * (Ss / 128), 256, 0, stream>>>(
        qB, kB, vB, relb + (size_t)l * RELN * Hh, ctxB);

    // Output projection (f32 out)
    gemm_bf16_wmma<false><<<gemm_grid(Mrow, Dm), 256, 0, stream>>>(
        ctxB, Wot, bo + (size_t)l * Dm, aoF, nullptr, Mrow, Dm, Dm);

    // LN1(x + attn_out)
    add_ln_kernel<<<Mrow, 256, 0, stream>>>(
        xf, aoF, ln1g + (size_t)l * Dm, ln1b + (size_t)l * Dm, yF, yB);

    // FFN
    gemm_bf16_wmma<true><<<gemm_grid(Mrow, DFF), 256, 0, stream>>>(
        yB, W1t, b1 + (size_t)l * DFF, nullptr, hB, Mrow, DFF, Dm);
    gemm_bf16_wmma<false><<<gemm_grid(Mrow, Dm), 256, 0, stream>>>(
        hB, W2t, b2 + (size_t)l * Dm, ffF, nullptr, Mrow, Dm, DFF);

    // LN2(y + ff) -> next layer's x
    add_ln_kernel<<<Mrow, 256, 0, stream>>>(
        yF, ffF, ln2g + (size_t)l * Dm, ln2b + (size_t)l * Dm, xf, xb);
  }

  // Mean pool + classifier head
  pool_kernel<<<(Bb * Dm + 255) / 256, 256, 0, stream>>>(xf, plF, plB);
  wconv_kernel<<<(Dm * (Dm / 2) + 255) / 256, 256, 0, stream>>>(cw1, C1t, Dm, Dm / 2);
  wconv_kernel<<<((Dm / 2) * NC + 255) / 256, 256, 0, stream>>>(cw2, C2t, Dm / 2, NC);
  gemm_bf16_wmma<true><<<gemm_grid(Bb, Dm / 2), 256, 0, stream>>>(
      plB, C1t, cb1, nullptr, h1B, Bb, Dm / 2, Dm);
  gemm_bf16_wmma<false><<<gemm_grid(Bb, NC), 256, 0, stream>>>(
      h1B, C2t, cb2, (float*)d_out, nullptr, Bb, NC, Dm / 2);
}
